// Attention_65446711657259
// MI455X (gfx1250) — matrix operand order repaired and checked by reading, not compiled, not run on hardware
//
#include <hip/hip_runtime.h>

// Problem constants (match reference)
#define BB 8
#define SS 2048
#define DD 512

typedef __attribute__((ext_vector_type(16))) _Float16 v16h;
typedef __attribute__((ext_vector_type(8)))  float    v8f;
typedef __attribute__((ext_vector_type(4)))  float    v4f;

union ABFrag {
    v16h h;
    v4f  f[2];   // two 16-byte pieces (8 halves each)
};

// Branch-free tanh: scores ~ N(0, ~0.65^2), clamp only guards the exp range.
__device__ __forceinline__ float tanh_fast(float x) {
    float xc = fminf(fmaxf(x, -10.0f), 10.0f);
    float e  = __expf(2.0f * xc);               // v_exp_f32, branchless
    return 1.0f - __fdividef(2.0f, e + 1.0f);   // v_rcp_f32, branchless
}

// ---------------------------------------------------------------------------
// Prep kernel: kh[b][i][d] = f16(q*Wd)   (row-major, K=d contiguous -> GEMM1 B)
//              qT[b][d][i] = f16(q)      (transposed, K=i contiguous -> GEMM2 B)
// grid = (DD/32, SS/32, BB), block = 256
// ---------------------------------------------------------------------------
__global__ __launch_bounds__(256) void prep_kernel(
    const float* __restrict__ q, const float* __restrict__ Wd,
    _Float16* __restrict__ kh, _Float16* __restrict__ qT)
{
    __shared__ _Float16 tile[32 * 33];   // padded to dodge bank conflicts
    const int tid = threadIdx.x;
    const int d0 = blockIdx.x * 32;
    const int i0 = blockIdx.y * 32;
    const int b  = blockIdx.z;

#pragma unroll
    for (int l = 0; l < 4; ++l) {
        int idx = tid + l * 256;            // 0..1023
        int ii  = idx >> 5;                 // i within tile
        int dd  = idx & 31;                 // d within tile
        size_t g = (size_t)(b * SS + i0 + ii) * DD + d0 + dd;
        float v = q[g];
        kh[g] = (_Float16)(v * Wd[d0 + dd]);
        tile[dd * 33 + ii] = (_Float16)v;
    }
    __syncthreads();
#pragma unroll
    for (int l = 0; l < 4; ++l) {
        int idx = tid + l * 256;
        int dd  = idx >> 5;
        int ii  = idx & 31;
        qT[(size_t)(b * DD + d0 + dd) * SS + i0 + ii] = tile[dd * 33 + ii];
    }
}

// ---------------------------------------------------------------------------
// Fused attention kernel. One workgroup = 32 j-rows of p, 8 waves (256 thr).
// LDS: sP   32x512  f16  (p tile, GEMM1 A source)   32 KB  (reused for sums)
//      sPr  32x2048 f16  (exp(sd), unnormalized)   128 KB
// Total 160 KB -> two workgroups per 320 KB WGP.
// grid = (SS/32, BB), block = 256, dyn smem = 163840 B
// ---------------------------------------------------------------------------
#define SMEM_BYTES (32 * DD * 2 + 32 * SS * 2)   // 163840

__global__ __launch_bounds__(256) void attn_kernel(
    const float* __restrict__ p, const float* __restrict__ vd,
    const _Float16* __restrict__ kh, const _Float16* __restrict__ qT,
    float* __restrict__ out)
{
    extern __shared__ __align__(16) char smem[];
    _Float16* sP   = (_Float16*)smem;                 // 32*512 halves
    _Float16* sPr  = (_Float16*)(smem + 32 * DD * 2); // 32*2048 halves
    float*    sSum = (float*)smem;                    // 32 floats, reused after phase 1

    const int tid  = threadIdx.x;
    const int b    = blockIdx.y;
    const int j0   = blockIdx.x * 32;
    const int wave = tid >> 5;        // 0..7
    const int lane = tid & 31;
    const int hi   = lane >> 4;       // lane half (0/1)
    const int lm   = lane & 15;

    // ---- Phase 0: stage p tile (32 x 512) to LDS as f16 ----
    for (int idx = tid; idx < 32 * DD; idx += 256) {
        int m = idx >> 9;             // DD == 512
        int d = idx & (DD - 1);
        sP[idx] = (_Float16)p[(size_t)(b * SS + j0 + m) * DD + d];
    }
    __syncthreads();

    // ---- Phase 1: scores = p x (q*Wd)^T, tanh*vd, exp -> f16 P in LDS ----
    // Each wave owns 16 i-tiles of 16 columns; two M-tiles (rows 0-15, 16-31).
    float part[2][8];
#pragma unroll
    for (int mt = 0; mt < 2; ++mt)
#pragma unroll
        for (int r = 0; r < 8; ++r) part[mt][r] = 0.0f;

    const _Float16* aRow0 = sP + (lm)      * DD + hi * 8;
    const _Float16* aRow1 = sP + (16 + lm) * DD + hi * 8;
    for (int it = 0; it < 16; ++it) {
        const int i0 = (it * 8 + wave) * 16;
        v8f acc0, acc1;
#pragma unroll
        for (int r = 0; r < 8; ++r) { acc0[r] = 0.0f; acc1[r] = 0.0f; }

        const _Float16* bRow = kh + (size_t)(b * SS + i0 + lm) * DD + hi * 8;
#pragma unroll 4
        for (int d0 = 0; d0 < DD; d0 += 32) {
            ABFrag a0, a1, bf;
            a0.f[0] = *(const v4f*)(aRow0 + d0);
            a0.f[1] = *(const v4f*)(aRow0 + d0 + 16);
            a1.f[0] = *(const v4f*)(aRow1 + d0);
            a1.f[1] = *(const v4f*)(aRow1 + d0 + 16);
            bf.f[0] = *(const v4f*)(bRow + d0);
            bf.f[1] = *(const v4f*)(bRow + d0 + 16);
            acc0 = __builtin_amdgcn_wmma_f32_16x16x32_f16(
                false, a0.h, false, bf.h, (short)0, acc0, false, false);
            acc1 = __builtin_amdgcn_wmma_f32_16x16x32_f16(
                false, a1.h, false, bf.h, (short)0, acc1, false, false);
        }

        const int icol = i0 + lm;               // C-layout: column fixed per lane
        const float vdv = vd[icol];
#pragma unroll
        for (int r = 0; r < 8; ++r) {
            // sd in [-0.05,0.05] -> exp() needs no max subtraction
            float e0 = __expf(tanh_fast(acc0[r]) * vdv);
            float e1 = __expf(tanh_fast(acc1[r]) * vdv);
            sPr[(r + hi * 8) * SS + icol]      = (_Float16)e0;
            sPr[(16 + r + hi * 8) * SS + icol] = (_Float16)e1;
            part[0][r] += e0;
            part[1][r] += e1;
        }
    }
    __syncthreads();   // all sP reads done; safe to reuse as sSum

    if (tid < 32) sSum[tid] = 0.0f;
    __syncthreads();

    // Reduce partial row sums across the 16 column-lanes, then LDS atomics.
#pragma unroll
    for (int mt = 0; mt < 2; ++mt)
#pragma unroll
        for (int r = 0; r < 8; ++r) {
            float v = part[mt][r];
#pragma unroll
            for (int m = 1; m < 16; m <<= 1) v += __shfl_xor(v, m, 32);
            if (lm == 0) atomicAdd(&sSum[mt * 16 + r + hi * 8], v);
        }
    __syncthreads();

    // ---- Phase 3: out = P x q (K = i = 2048). Wave w owns d-cols w*64..+63 ----
    v8f acc2[2][4];
#pragma unroll
    for (int mt = 0; mt < 2; ++mt)
#pragma unroll
        for (int nt = 0; nt < 4; ++nt)
#pragma unroll
            for (int r = 0; r < 8; ++r) acc2[mt][nt][r] = 0.0f;

    const _Float16* pRow0 = sPr + (lm)      * SS + hi * 8;
    const _Float16* pRow1 = sPr + (16 + lm) * SS + hi * 8;
    for (int i0 = 0; i0 < SS; i0 += 32) {
        ABFrag a0, a1;
        a0.f[0] = *(const v4f*)(pRow0 + i0);
        a0.f[1] = *(const v4f*)(pRow0 + i0 + 16);
        a1.f[0] = *(const v4f*)(pRow1 + i0);
        a1.f[1] = *(const v4f*)(pRow1 + i0 + 16);
#pragma unroll
        for (int nt = 0; nt < 4; ++nt) {
            const int n = wave * 64 + nt * 16 + lm;   // output d column
            const _Float16* bp = qT + (size_t)(b * DD + n) * SS + i0 + hi * 8;
            ABFrag bf;
            bf.f[0] = *(const v4f*)bp;
            bf.f[1] = *(const v4f*)(bp + 16);
            acc2[0][nt] = __builtin_amdgcn_wmma_f32_16x16x32_f16(
                false, a0.h, false, bf.h, (short)0, acc2[0][nt], false, false);
            acc2[1][nt] = __builtin_amdgcn_wmma_f32_16x16x32_f16(
                false, a1.h, false, bf.h, (short)0, acc2[1][nt], false, false);
        }
    }

    // Epilogue: fold in 1/rowsum, store f32 (coalesced along d)
    float inv[2][8];
#pragma unroll
    for (int mt = 0; mt < 2; ++mt)
#pragma unroll
        for (int r = 0; r < 8; ++r)
            inv[mt][r] = __fdividef(1.0f, sSum[mt * 16 + r + hi * 8]);
#pragma unroll
    for (int mt = 0; mt < 2; ++mt)
#pragma unroll
        for (int nt = 0; nt < 4; ++nt) {
            const int d = wave * 64 + nt * 16 + lm;
#pragma unroll
            for (int r = 0; r < 8; ++r) {
                out[(size_t)(b * SS + j0 + mt * 16 + r + hi * 8) * DD + d] =
                    acc2[mt][nt][r] * inv[mt][r];
            }
        }
}

// ---------------------------------------------------------------------------
extern "C" void kernel_launch(void* const* d_in, const int* in_sizes, int n_in,
                              void* d_out, int out_size, void* d_ws, size_t ws_size,
                              hipStream_t stream) {
    const float* q  = (const float*)d_in[0];
    const float* p  = (const float*)d_in[1];
    const float* Wd = (const float*)d_in[2];
    const float* vd = (const float*)d_in[3];
    float* out = (float*)d_out;

    _Float16* kh = (_Float16*)d_ws;                        // 16 MB
    _Float16* qT = kh + (size_t)BB * SS * DD;              // 16 MB

    static bool attr_set = false;
    if (!attr_set) {
        hipFuncSetAttribute((const void*)attn_kernel,
                            hipFuncAttributeMaxDynamicSharedMemorySize, SMEM_BYTES);
        attr_set = true;
    }

    dim3 gPrep(DD / 32, SS / 32, BB);
    prep_kernel<<<gPrep, 256, 0, stream>>>(q, Wd, kh, qT);

    dim3 gAttn(SS / 32, BB);
    attn_kernel<<<gAttn, 256, SMEM_BYTES, stream>>>(p, vd, kh, qT, out);
}
